// GNN_46523085750274
// MI455X (gfx1250) — matrix operand order repaired and checked by reading, not compiled, not run on hardware
//
#include <hip/hip_runtime.h>
#include <hip/hip_bf16.h>

// ---------------- problem constants ----------------
#define NN    20000      // atom nodes
#define NLG   40000      // line-graph nodes
#define EE    80000      // directed edges main graph
#define ELG   200000     // line-graph edges
#define LL    5          // layers
#define DD    300        // emb dim
#define DD2   600        // hidden dim
#define ROWS  (NN + NLG) // 60000 rows per output slice

// GEMM padding (K multiple of 32 for wmma bf16, N multiple of 16)
#define KP1   320        // padded K for gemm1 (D=300 -> 320)
#define NP1   608        // padded N for gemm1 out / K for gemm2 (600 -> 608)
#define KT1   10         // 320/32
#define NT1   38         // 608/16
#define KT2   19         // 608/32
#define NT2   19         // 304/16 (N=300 padded to 304, stores masked)

#define EPB   8          // edges per block in scatter kernels
#define CDIV(a,b) (((a)+(b)-1)/(b))

typedef __attribute__((ext_vector_type(16))) __bf16 v16bf;
typedef __attribute__((ext_vector_type(8)))  float  v8f;

union Frag16 { v16bf v; uint4 q[2]; };

__device__ __forceinline__ unsigned short f2bf(float f) {
    unsigned int u = __float_as_uint(f);
    u += 0x7FFFu + ((u >> 16) & 1u);   // round-to-nearest-even
    return (unsigned short)(u >> 16);
}

// ---------------- elementwise / embedding kernels ----------------

// out[r][c] = t1[idx[r][0]][c] + t2[idx[r][1]][c]
__global__ void embed_add(const int* __restrict__ idx, const float* __restrict__ t1,
                          const float* __restrict__ t2, float* __restrict__ out, int M) {
    int gid = blockIdx.x * blockDim.x + threadIdx.x;
    if (gid >= M * DD) return;
    int r = gid / DD, c = gid - r * DD;
    out[gid] = t1[idx[2 * r] * DD + c] + t2[idx[2 * r + 1] * DD + c];
}

// aggr[r][c] = xn[r][c] + sl1[c] + sl2[c]   (self-loop message)
__global__ void aggr_init(const float* __restrict__ xn, const float* __restrict__ sl1,
                          const float* __restrict__ sl2, float* __restrict__ aggr, int M) {
    int gid = blockIdx.x * blockDim.x + threadIdx.x;
    if (gid >= M * DD) return;
    int c = gid % DD;
    aggr[gid] = xn[gid] + sl1[c] + sl2[c];
}

__global__ void zero_f32(float* __restrict__ p, int n) {
    int t = blockIdx.x * blockDim.x + threadIdx.x;
    if (t < n) p[t] = 0.f;
}

// ---------------- scatter (message passing) kernels ----------------
// Block = 320 threads; thread c<300 owns channel c; EPB edges per block.

// main graph, layer 0: ea from bond/dir embedding tables
__global__ void scatter_main_l0(const float* __restrict__ h, const int* __restrict__ ei,
                                const int* __restrict__ eattr, const float* __restrict__ emb1,
                                const float* __restrict__ emb2, float* __restrict__ aggr) {
    int c = threadIdx.x; if (c >= DD) return;
    int e0 = blockIdx.x * EPB;
    for (int k = 0; k < EPB; ++k) {
        int j = e0 + k; if (j >= EE) break;
        int src = ei[j], dst = ei[EE + j];
        int a0 = eattr[2 * j], a1 = eattr[2 * j + 1];
        float v = h[(size_t)src * DD + c] + emb1[a0 * DD + c] + emb2[a1 * DD + c];
        atomicAdd(&aggr[(size_t)dst * DD + c], v);
    }
}

// main graph, layer>0: edges rewired from lg map, ea = e_prev (duplicated)
__global__ void scatter_main_rw(const float* __restrict__ h, const float* __restrict__ e_prev,
                                const int* __restrict__ lgmap, float* __restrict__ aggr) {
    int c = threadIdx.x; if (c >= DD) return;
    int e0 = blockIdx.x * EPB;
    for (int k = 0; k < EPB; ++k) {
        int j = e0 + k; if (j >= 2 * NLG) break;
        int jm = (j < NLG) ? j : j - NLG;
        int src, dst;
        if (j < NLG) { src = lgmap[jm];       dst = lgmap[NLG + jm]; }
        else         { src = lgmap[NLG + jm]; dst = lgmap[jm]; }
        float v = h[(size_t)src * DD + c] + e_prev[(size_t)jm * DD + c];
        atomicAdd(&aggr[(size_t)dst * DD + c], v);
    }
}

// line graph, layer 0: ea from atom/chirality tables via map2 -> x
__global__ void scatter_lg_l0(const float* __restrict__ e, const int* __restrict__ lgei,
                              const int* __restrict__ map2, const int* __restrict__ x,
                              const float* __restrict__ emb1, const float* __restrict__ emb2,
                              float* __restrict__ aggr) {
    int c = threadIdx.x; if (c >= DD) return;
    int e0 = blockIdx.x * EPB;
    for (int k = 0; k < EPB; ++k) {
        int j = e0 + k; if (j >= ELG) break;
        int src = lgei[j], dst = lgei[ELG + j];
        int nd = map2[j];
        int a0 = x[2 * nd], a1 = x[2 * nd + 1];
        float v = e[(size_t)src * DD + c] + emb1[a0 * DD + c] + emb2[a1 * DD + c];
        atomicAdd(&aggr[(size_t)dst * DD + c], v);
    }
}

// line graph, layer>0: ea = h_prev[map2]
__global__ void scatter_lg(const float* __restrict__ e, const int* __restrict__ lgei,
                           const int* __restrict__ map2, const float* __restrict__ h_prev,
                           float* __restrict__ aggr) {
    int c = threadIdx.x; if (c >= DD) return;
    int e0 = blockIdx.x * EPB;
    for (int k = 0; k < EPB; ++k) {
        int j = e0 + k; if (j >= ELG) break;
        int src = lgei[j], dst = lgei[ELG + j];
        float v = e[(size_t)src * DD + c] + h_prev[(size_t)map2[j] * DD + c];
        atomicAdd(&aggr[(size_t)dst * DD + c], v);
    }
}

// ---------------- precision conversion / weight packing ----------------

// f32 [M x K] row-major -> bf16 [M x KP] row-major, zero-padded K
__global__ void f32_to_bf16_pad(const float* __restrict__ src, unsigned short* __restrict__ dst,
                                int M, int K, int KP) {
    int gid = blockIdx.x * blockDim.x + threadIdx.x;
    if (gid >= M * KP) return;
    int r = gid / KP, k = gid - r * KP;
    float v = (k < K) ? src[(size_t)r * K + k] : 0.f;
    dst[gid] = f2bf(v);
}

// Pack W [K x Nw] f32 row-major into WMMA B-fragment layout:
// out[((kt*NT + nt)*32 + lane)*8 + v] holds bf16 pair
//   B[k0 + half*16 + 2v][n0+n], B[k0 + half*16 + 2v + 1][n0+n]
// with half = lane>>4, n = lane&15.  Launch: <<<KT*NT, 256>>>.
__global__ void pack_b(const float* __restrict__ W, unsigned int* __restrict__ out,
                       int K, int Nw, int NT) {
    int t = threadIdx.x;
    int v = t & 7, lane = t >> 3;
    int tile = blockIdx.x;
    int nt = tile % NT, kt = tile / NT;
    int half = lane >> 4, n = lane & 15;
    int col = nt * 16 + n;
    int k = kt * 32 + half * 16 + 2 * v;
    unsigned short lo = (k < K && col < Nw)     ? f2bf(W[(size_t)k * Nw + col])       : 0;
    unsigned short hi = (k + 1 < K && col < Nw) ? f2bf(W[(size_t)(k + 1) * Nw + col]) : 0;
    out[(size_t)tile * 256 + t] = (unsigned)lo | ((unsigned)hi << 16);
}

// ---------------- WMMA GEMM: C = relu?(A@B + bias) ----------------
// A: bf16 [Mtiles*16 x KP] row-major; Bpack: packed per pack_b.
// One wave per 16x16 output tile.  BF16OUT: store bf16 (ldc=NP1, all cols);
// else store f32 only for col < Nvalid.
template<int BF16OUT, int RELU>
__global__ void __launch_bounds__(256)
wmma_gemm(const unsigned short* __restrict__ A, int KP,
          const unsigned int* __restrict__ Bp,
          const float* __restrict__ bias, int Nvalid,
          int KT, int NT, void* __restrict__ Cout, int ldc, int Mtiles) {
    int wid = (int)((blockIdx.x * blockDim.x + threadIdx.x) >> 5);
    if (wid >= Mtiles * NT) return;             // whole-wave uniform exit
    int mt = wid / NT, nt = wid - mt * NT;
    int lane = threadIdx.x & 31;
    int half = lane >> 4, l16 = lane & 15;

    const unsigned short* arow = A + (size_t)(mt * 16 + l16) * KP;
    v8f acc = {};
    for (int kt = 0; kt < KT; ++kt) {
        Frag16 a, b;
        int k0 = kt * 32;
        a.q[0] = *(const uint4*)(arow + k0 + half * 16);
        a.q[1] = *(const uint4*)(arow + k0 + half * 16 + 8);
        const unsigned int* bp = Bp + (((size_t)kt * NT + nt) * 32 + lane) * 8;
        b.q[0] = *(const uint4*)(bp);
        b.q[1] = *(const uint4*)(bp + 4);
        acc = __builtin_amdgcn_wmma_f32_16x16x32_bf16(
            false, a.v, false, b.v, (short)0, acc, false, false);
    }
    int col = nt * 16 + l16;
    float bv = (col < Nvalid) ? bias[col] : 0.f;
    if (BF16OUT) {
        unsigned short* C = (unsigned short*)Cout;
#pragma unroll
        for (int r = 0; r < 8; ++r) {
            float v = acc[r] + bv;
            if (RELU) v = fmaxf(v, 0.f);
            C[(size_t)(mt * 16 + half * 8 + r) * ldc + col] = f2bf(v);
        }
    } else {
        if (col < Nvalid) {
            float* C = (float*)Cout;
#pragma unroll
            for (int r = 0; r < 8; ++r) {
                float v = acc[r] + bv;
                if (RELU) v = fmaxf(v, 0.f);
                C[(size_t)(mt * 16 + half * 8 + r) * ldc + col] = v;
            }
        }
    }
}

// ---------------- batchnorm ----------------

__global__ void bn_stats(const float* __restrict__ h, float* __restrict__ sum,
                         float* __restrict__ sumsq, int M, int rpb) {
    int c = threadIdx.x; if (c >= DD) return;
    int r0 = blockIdx.x * rpb;
    float s = 0.f, s2 = 0.f;
    for (int i = 0; i < rpb; ++i) {
        int r = r0 + i; if (r >= M) break;
        float v = h[(size_t)r * DD + c];
        s += v; s2 += v * v;
    }
    atomicAdd(&sum[c], s);
    atomicAdd(&sumsq[c], s2);
}

__global__ void bn_apply(float* __restrict__ h, const float* __restrict__ sum,
                         const float* __restrict__ sumsq, const float* __restrict__ gamma,
                         const float* __restrict__ beta, int M, int relu) {
    int gid = blockIdx.x * blockDim.x + threadIdx.x;
    if (gid >= M * DD) return;
    int c = gid % DD;
    float inv = 1.f / (float)M;
    float mu  = sum[c] * inv;
    float var = sumsq[c] * inv - mu * mu;
    float v = (h[gid] - mu) * rsqrtf(var + 1e-5f) * gamma[c] + beta[c];
    if (relu) v = fmaxf(v, 0.f);
    h[gid] = v;
}

// ---------------- host-side helpers ----------------

static void mlp_bn(hipStream_t s, int M, float* aggr, unsigned short* abf, unsigned short* t1,
                   const float* w1, const float* b1, const float* w2, const float* b2,
                   const float* gamma, const float* beta, int relu,
                   unsigned int* w1p, unsigned int* w2p, float* stats, float* hout) {
    // aggr f32 -> bf16 padded
    f32_to_bf16_pad<<<CDIV(M * KP1, 256), 256, 0, s>>>(aggr, abf, M, DD, KP1);
    // pack weights into WMMA B layout
    pack_b<<<KT1 * NT1, 256, 0, s>>>(w1, w1p, DD, DD2, NT1);
    pack_b<<<KT2 * NT2, 256, 0, s>>>(w2, w2p, DD2, DD, NT2);
    // GEMM1: [M x 300] @ [300 x 600] + b1, ReLU -> bf16 [M x 608]
    int mt = M / 16;
    int waves1 = mt * NT1;
    wmma_gemm<1, 1><<<CDIV(waves1 * 32, 256), 256, 0, s>>>(
        abf, KP1, w1p, b1, DD2, KT1, NT1, (void*)t1, NP1, mt);
    // GEMM2: [M x 600] @ [600 x 300] + b2 -> f32 [M x 300] (into d_out slice)
    int waves2 = mt * NT2;
    wmma_gemm<0, 0><<<CDIV(waves2 * 32, 256), 256, 0, s>>>(
        t1, NP1, w2p, b2, DD, KT2, NT2, (void*)hout, DD, mt);
    // batchnorm (training mode, biased variance) + optional ReLU, in place
    zero_f32<<<1, 2 * DD, 0, s>>>(stats, 2 * DD);
    const int rpb = 64;
    bn_stats<<<CDIV(M, rpb), 320, 0, s>>>(hout, stats, stats + DD, M, rpb);
    bn_apply<<<CDIV(M * DD, 256), 256, 0, s>>>(hout, stats, stats + DD, gamma, beta, M, relu);
}

extern "C" void kernel_launch(void* const* d_in, const int* in_sizes, int n_in,
                              void* d_out, int out_size, void* d_ws, size_t ws_size,
                              hipStream_t stream) {
    const int*   x        = (const int*)d_in[0];
    const int*   ei       = (const int*)d_in[1];
    const int*   eattr    = (const int*)d_in[2];
    const int*   lg_x     = (const int*)d_in[3];
    const int*   lgei     = (const int*)d_in[4];
    const int*   lgmap    = (const int*)d_in[5];
    const int*   map2     = (const int*)d_in[6];
    const float* node_e1  = (const float*)d_in[7];
    const float* node_e2  = (const float*)d_in[8];
    const float* ge1      = (const float*)d_in[9];
    const float* ge2      = (const float*)d_in[10];
    const float* g_emb1   = (const float*)d_in[11];  // [L,6,D]
    const float* g_emb2   = (const float*)d_in[12];  // [L,3,D]
    const float* g_w1     = (const float*)d_in[13];  // [L,D,2D]
    const float* g_b1     = (const float*)d_in[14];
    const float* g_w2     = (const float*)d_in[15];
    const float* g_b2     = (const float*)d_in[16];
    const float* g_gamma  = (const float*)d_in[17];
    const float* g_beta   = (const float*)d_in[18];
    const float* lg_emb1  = (const float*)d_in[19];  // [L,120,D]
    const float* lg_emb2  = (const float*)d_in[20];  // [L,3,D]
    const float* lg_w1    = (const float*)d_in[21];
    const float* lg_b1    = (const float*)d_in[22];
    const float* lg_w2    = (const float*)d_in[23];
    const float* lg_b2    = (const float*)d_in[24];
    const float* lg_gamma = (const float*)d_in[25];
    const float* lg_beta  = (const float*)d_in[26];
    float* out = (float*)d_out;   // [6, 60000, 300]: h_list then e_list per slice

    // workspace carve (~123 MB)
    char* ws = (char*)d_ws;
    size_t o = 0;
    float*          aggr = (float*)(ws + o);          o += (size_t)NLG * DD * sizeof(float);
    unsigned short* abf  = (unsigned short*)(ws + o); o += (size_t)NLG * KP1 * 2;
    unsigned short* t1   = (unsigned short*)(ws + o); o += (size_t)NLG * NP1 * 2;
    unsigned int*   w1p  = (unsigned int*)(ws + o);   o += (size_t)KT1 * NT1 * 256 * 4;
    unsigned int*   w2p  = (unsigned int*)(ws + o);   o += (size_t)KT2 * NT2 * 256 * 4;
    float*          stats= (float*)(ws + o);          o += 2 * DD * sizeof(float);
    (void)ws_size; (void)in_sizes; (void)n_in; (void)out_size;

    // ----- initial embeddings: h0, e0 directly into output slice 0 -----
    embed_add<<<CDIV(NN * DD, 256), 256, 0, stream>>>(x, node_e1, node_e2, out, NN);
    embed_add<<<CDIV(NLG * DD, 256), 256, 0, stream>>>(lg_x, ge1, ge2, out + (size_t)NN * DD, NLG);

    for (int layer = 0; layer < LL; ++layer) {
        const float* h_in  = out + (size_t)layer * ROWS * DD;       // h_list[layer]
        const float* e_in  = h_in + (size_t)NN * DD;                // e_list[layer]
        float*       h_out = out + (size_t)(layer + 1) * ROWS * DD; // h_list[layer+1]
        float*       e_out = h_out + (size_t)NN * DD;               // e_list[layer+1]

        // ===== main-graph GINConv =====
        const float* sl1 = g_emb1 + ((size_t)layer * 6 + 4) * DD;   // sl_idx = 4
        const float* sl2 = g_emb2 + (size_t)layer * 3 * DD;         // row 0
        aggr_init<<<CDIV(NN * DD, 256), 256, 0, stream>>>(h_in, sl1, sl2, aggr, NN);
        if (layer == 0)
            scatter_main_l0<<<CDIV(EE, EPB), 320, 0, stream>>>(h_in, ei, eattr, g_emb1, g_emb2, aggr);
        else
            scatter_main_rw<<<CDIV(2 * NLG, EPB), 320, 0, stream>>>(h_in, e_in, lgmap, aggr);
        mlp_bn(stream, NN, aggr, abf, t1,
               g_w1 + (size_t)layer * DD * DD2, g_b1 + (size_t)layer * DD2,
               g_w2 + (size_t)layer * DD2 * DD, g_b2 + (size_t)layer * DD,
               g_gamma + (size_t)layer * DD, g_beta + (size_t)layer * DD,
               (layer < LL - 1) ? 1 : 0, w1p, w2p, stats, h_out);

        // ===== line-graph GINConv =====
        const float* lsl1 = lg_emb1 + ((size_t)layer * 120 + 119) * DD;  // sl_idx = 119
        const float* lsl2 = lg_emb2 + (size_t)layer * 3 * DD;            // row 0
        aggr_init<<<CDIV(NLG * DD, 256), 256, 0, stream>>>(e_in, lsl1, lsl2, aggr, NLG);
        if (layer == 0)
            scatter_lg_l0<<<CDIV(ELG, EPB), 320, 0, stream>>>(e_in, lgei, map2, x, lg_emb1, lg_emb2, aggr);
        else
            scatter_lg<<<CDIV(ELG, EPB), 320, 0, stream>>>(e_in, lgei, map2, h_in, aggr);
        mlp_bn(stream, NLG, aggr, abf, t1,
               lg_w1 + (size_t)layer * DD * DD2, lg_b1 + (size_t)layer * DD2,
               lg_w2 + (size_t)layer * DD2 * DD, lg_b2 + (size_t)layer * DD,
               lg_gamma + (size_t)layer * DD, lg_beta + (size_t)layer * DD,
               (layer < LL - 1) ? 1 : 0, w1p, w2p, stats, e_out);
    }
}
